// SelfAttention_80771154969227
// MI455X (gfx1250) — matrix operand order repaired and checked by reading, not compiled, not run on hardware
//
#include <hip/hip_runtime.h>
#include <hip/hip_bf16.h>

typedef __attribute__((ext_vector_type(16))) _Float16 v16h;
typedef __attribute__((ext_vector_type(8)))  _Float16 v8h;
typedef __attribute__((ext_vector_type(4)))  _Float16 v4h;
typedef __attribute__((ext_vector_type(8)))  float    v8f;
typedef __attribute__((ext_vector_type(4)))  float    v4f;

#define WMMA_F16(a, b, c) \
    __builtin_amdgcn_wmma_f32_16x16x32_f16(false, (a), false, (b), (short)0, (c), false, false)

// ---------------------------------------------------------------------------
// Kernel 0: convert the 4 weight matrices (each 256x256 fp32, torch [out,in])
// to f16, packed [Wq|Wk|Wv|Wp].  Row-major [e][d]: a B-fragment column is a
// contiguous 32-byte run of K values per lane.
// ---------------------------------------------------------------------------
__global__ __launch_bounds__(256) void k_convw(const float* __restrict__ Wq,
                                               const float* __restrict__ Wk,
                                               const float* __restrict__ Wv,
                                               const float* __restrict__ Wp,
                                               _Float16* __restrict__ W16) {
    int id = blockIdx.x * 256 + threadIdx.x;       // 0 .. 262143
    int iw = id >> 16, off = id & 65535;
    const float* s = (iw == 0) ? Wq : (iw == 1) ? Wk : (iw == 2) ? Wv : Wp;
    W16[id] = (_Float16)s[off];
}

// ---------------------------------------------------------------------------
// Kernel 1: m = sigmoid(mask_logits); pass-through copy of mask_logits.
// ---------------------------------------------------------------------------
__global__ __launch_bounds__(256) void k_mask(const float* __restrict__ ml,
                                              float* __restrict__ m_mul,
                                              float* __restrict__ out_mask) {
    int id = blockIdx.x * 256 + threadIdx.x;       // 0 .. 32767
    float x = ml[id];
    m_mul[id]    = 1.0f / (1.0f + __expf(-x));
    out_mask[id] = x;
}

// ---------------------------------------------------------------------------
// Kernel 2: FUSED qkv-projection + attention.  One block per (b,h).
//   phase 0: X[b] 64x256 f32 -> f16 LDS (L2-resident re-read; no HBM qkv
//            round trip)
//   phase 1: per-wave q/k/v head-slice projection on WMMA (A-frags hoisted to
//            registers, reused across 3 weights x 2 N-tiles), results staged
//            to Qs/Ks/Vt LDS in WMMA operand layouts
//   phase 2: QK^T (4 WMMA) -> scale*mask+log(mask), *graph (direct coalesced
//            global gathers, L2/WGP$-resident) -> wave32 xor-shuffle softmax
//            -> f32 att out + f16 restage -> P@V (4 WMMA) -> merged f16 y.
// ---------------------------------------------------------------------------
__global__ __launch_bounds__(128) void k_attn_fused(
        const float* __restrict__ X,
        const _Float16* __restrict__ W16,
        const float* __restrict__ bq,
        const float* __restrict__ bk,
        const float* __restrict__ bv,
        const float* __restrict__ graph,
        const float* __restrict__ m_mul,
        float* __restrict__ out_att,
        _Float16* __restrict__ y16) {
    __shared__ _Float16 Xs[64][264];       // 33792 B
    __shared__ _Float16 Qs[64][40];        //  5120 B
    __shared__ _Float16 Ks[64][40];        //  5120 B
    __shared__ _Float16 Vt[32][72];        //  4608 B   Vt[d][k]
    __shared__ _Float16 Ps[4][16][72];     //  9216 B   total 57856 B

    const int t  = threadIdx.x;
    const int bh = blockIdx.x;             // 0..8191
    const int b  = bh >> 3, h = bh & 7;

    // ---- phase 0: stage X[b] fp32 -> f16 ----
    const float* Xb = X + (size_t)b * 16384;
#pragma unroll
    for (int i = 0; i < 32; ++i) {
        int flat = (i * 128 + t) * 4;              // 16384 floats total
        int row = flat >> 8, col = flat & 255;
        v4f x = *(const v4f*)(Xb + flat);
        v4h hv;
        hv[0] = (_Float16)x[0]; hv[1] = (_Float16)x[1];
        hv[2] = (_Float16)x[2]; hv[3] = (_Float16)x[3];
        *(v4h*)&Xs[row][col] = hv;
    }
    __syncthreads();

    const int wave = t >> 5, lane = t & 31, laneN = lane & 15;
    const bool hi = lane >= 16;
    const int m0 = wave * 16;
    const int alo = hi ? 16 : 0;

    // ---- hoist all 8 A-fragments (rows m0..m0+15, K=0..255) ----
    v16h A[8];
#pragma unroll
    for (int kk = 0; kk < 8; ++kk) {
        int k0 = kk * 32;
        *(v8h*)&A[kk]       = *(const v8h*)&Xs[m0 + laneN][k0 + alo];
        *((v8h*)&A[kk] + 1) = *(const v8h*)&Xs[m0 + laneN][k0 + 8 + alo];
    }

    // ---- phase 1: q/k/v head-slice projection ----
#pragma unroll
    for (int w = 0; w < 3; ++w) {
        const _Float16* Wb = W16 + (size_t)w * 65536;
        const float* bias  = (w == 0) ? bq : (w == 1) ? bk : bv;
#pragma unroll
        for (int nt = 0; nt < 2; ++nt) {
            const int e = h * 32 + nt * 16 + laneN;    // output column
            v8f acc = {};
#pragma unroll
            for (int kk = 0; kk < 8; ++kk) {
                int k0 = kk * 32;
                v16h bf;
                const _Float16* bp_ = Wb + (size_t)e * 256 + k0 + (hi ? 16 : 0);
                *(v8h*)&bf       = *(const v8h*)(bp_);
                *((v8h*)&bf + 1) = *(const v8h*)(bp_ + 8);
                acc = WMMA_F16(A[kk], bf, acc);
            }
            const float bv_ = bias[e];
            const int d = nt * 16 + laneN;             // 0..31 within head
#pragma unroll
            for (int r = 0; r < 8; ++r) {
                int row = m0 + r + (hi ? 8 : 0);
                _Float16 val = (_Float16)(acc[r] + bv_);
                if (w == 0)      Qs[row][d] = val;
                else if (w == 1) Ks[row][d] = val;
                else             Vt[d][row] = val;
            }
        }
    }
    __syncthreads();

    // ---- phase 2a: S = Q (16x32) @ K^T (32x64) ----
    v16h aq;
    *(v8h*)&aq       = *(const v8h*)&Qs[m0 + laneN][alo];
    *((v8h*)&aq + 1) = *(const v8h*)&Qs[m0 + laneN][8 + alo];
    v8f S[4];
#pragma unroll
    for (int nt = 0; nt < 4; ++nt) {
        v16h bf;                                       // B[k][n] = K[n][k]
        const _Float16* kp = &Ks[nt * 16 + laneN][hi ? 16 : 0];
        *(v8h*)&bf       = *(const v8h*)(kp);
        *((v8h*)&bf + 1) = *(const v8h*)(kp + 8);
        v8f z = {};
        S[nt] = WMMA_F16(aq, bf, z);
    }

    // ---- phase 2b: scale, sigmoid-mask mul + log, graph gate ----
    const float scale = 0.17677669529663687f;          // 1/sqrt(32)
    const float* gb  = graph + (size_t)b * 4096;
    const float* mmb = m_mul + (size_t)h * 4096;
#pragma unroll
    for (int nt = 0; nt < 4; ++nt) {
        int col = nt * 16 + laneN;
#pragma unroll
        for (int r = 0; r < 8; ++r) {
            int row = m0 + r + (hi ? 8 : 0);
            float m = mmb[row * 64 + col];             // 2x64B segments/instr
            float g = gb[row * 64 + col];
            float s = S[nt][r] * scale * m + __logf(m + 1e-8f);
            S[nt][r] = s * g;
        }
    }

    // ---- phase 2c: softmax over 64 cols ----
#pragma unroll
    for (int r = 0; r < 8; ++r) {
        float mx = fmaxf(fmaxf(S[0][r], S[1][r]), fmaxf(S[2][r], S[3][r]));
        mx = fmaxf(mx, __shfl_xor(mx, 1, 32));
        mx = fmaxf(mx, __shfl_xor(mx, 2, 32));
        mx = fmaxf(mx, __shfl_xor(mx, 4, 32));
        mx = fmaxf(mx, __shfl_xor(mx, 8, 32));
        float p0 = __expf(S[0][r] - mx), p1 = __expf(S[1][r] - mx);
        float p2 = __expf(S[2][r] - mx), p3 = __expf(S[3][r] - mx);
        float sum = (p0 + p1) + (p2 + p3);
        sum += __shfl_xor(sum, 1, 32);
        sum += __shfl_xor(sum, 2, 32);
        sum += __shfl_xor(sum, 4, 32);
        sum += __shfl_xor(sum, 8, 32);
        float inv = 1.0f / sum;
        S[0][r] = p0 * inv; S[1][r] = p1 * inv;
        S[2][r] = p2 * inv; S[3][r] = p3 * inv;
    }

    // ---- phase 2d: att out (f32) + f16 restage ----
    float* attb = out_att + (size_t)bh * 4096;
#pragma unroll
    for (int nt = 0; nt < 4; ++nt) {
        int col = nt * 16 + laneN;
#pragma unroll
        for (int r = 0; r < 8; ++r) {
            int rl = r + (hi ? 8 : 0);
            attb[(size_t)(m0 + rl) * 64 + col] = S[nt][r];
            Ps[wave][rl][col] = (_Float16)S[nt][r];    // wave-private region
        }
    }

    // ---- phase 2e: Y = P (16x64) @ V (64x32) ----
    v8f Y0 = {}, Y1 = {};
#pragma unroll
    for (int k0 = 0; k0 < 64; k0 += 32) {
        v16h pa;
        *(v8h*)&pa       = *(const v8h*)&Ps[wave][laneN][k0 + alo];
        *((v8h*)&pa + 1) = *(const v8h*)&Ps[wave][laneN][k0 + 8 + alo];
        v16h vb0, vb1;                                 // B[k][d] = Vt[d][k]
        const _Float16* vp0 = &Vt[laneN][k0 + (hi ? 16 : 0)];
        const _Float16* vp1 = &Vt[16 + laneN][k0 + (hi ? 16 : 0)];
        *(v8h*)&vb0       = *(const v8h*)(vp0);
        *((v8h*)&vb0 + 1) = *(const v8h*)(vp0 + 8);
        *(v8h*)&vb1       = *(const v8h*)(vp1);
        *((v8h*)&vb1 + 1) = *(const v8h*)(vp1 + 8);
        Y0 = WMMA_F16(pa, vb0, Y0);
        Y1 = WMMA_F16(pa, vb1, Y1);
    }
#pragma unroll
    for (int r = 0; r < 8; ++r) {
        int lAg = m0 + r + (hi ? 8 : 0);
        size_t base = ((size_t)(b * 64 + lAg)) * 256 + h * 32;
        y16[base + laneN]      = (_Float16)Y0[r];
        y16[base + 16 + laneN] = (_Float16)Y1[r];
    }
}

// ---------------------------------------------------------------------------
// Kernel 3: output projection.  y16 tile staged via CDNA5 async-to-LDS copy
// (ASYNCcnt-tracked, bypasses VGPRs), A-fragments hoisted to registers.
// ---------------------------------------------------------------------------
__global__ __launch_bounds__(128) void k_proj(const _Float16* __restrict__ y16,
                                              const _Float16* __restrict__ Wp16,
                                              const float* __restrict__ bp,
                                              float* __restrict__ out_y) {
    __shared__ _Float16 Ys[64][264];
    const int t = threadIdx.x;
    const int m0blk = blockIdx.x * 64;
    const _Float16* src = y16 + (size_t)m0blk * 256;

    // 64x256 halves = 16384; 8 iters x 128 lanes x (2 x b128) = 32 KB
#pragma unroll
    for (int i = 0; i < 8; ++i) {
        int flat = (i * 128 + t) * 16;
        int row = flat >> 8, col = flat & 255;
        unsigned lds0 = (unsigned)(size_t)&Ys[row][col];
        unsigned lds1 = (unsigned)(size_t)&Ys[row][col + 8];
        unsigned long long g0 = (unsigned long long)(size_t)(src + flat);
        unsigned long long g1 = (unsigned long long)(size_t)(src + flat + 8);
        asm volatile("global_load_async_to_lds_b128 %0, %1, off"
                     :: "v"(lds0), "v"(g0) : "memory");
        asm volatile("global_load_async_to_lds_b128 %0, %1, off"
                     :: "v"(lds1), "v"(g1) : "memory");
    }
    asm volatile("s_wait_asynccnt 0x0" ::: "memory");
    __syncthreads();

    const int wave = t >> 5, lane = t & 31, laneN = lane & 15;
    const bool hi = lane >= 16;
    const int m0 = wave * 16;
    const int alo = hi ? 16 : 0;

    v16h A[8];
#pragma unroll
    for (int kk = 0; kk < 8; ++kk) {
        int k0 = kk * 32;
        *(v8h*)&A[kk]       = *(const v8h*)&Ys[m0 + laneN][k0 + alo];
        *((v8h*)&A[kk] + 1) = *(const v8h*)&Ys[m0 + laneN][k0 + 8 + alo];
    }

#pragma unroll
    for (int nt = 0; nt < 16; ++nt) {
        const int e = nt * 16 + laneN;
        v8f acc = {};
#pragma unroll
        for (int kk = 0; kk < 8; ++kk) {
            int k0 = kk * 32;
            v16h bf;
            const _Float16* bp_ = Wp16 + (size_t)e * 256 + k0 + (hi ? 16 : 0);
            *(v8h*)&bf       = *(const v8h*)(bp_);
            *((v8h*)&bf + 1) = *(const v8h*)(bp_ + 8);
            acc = WMMA_F16(A[kk], bf, acc);
        }
        const float bv_ = bp[e];
#pragma unroll
        for (int r = 0; r < 8; ++r) {
            int g = m0blk + m0 + r + (hi ? 8 : 0);
            out_y[(size_t)g * 256 + e] = acc[r] + bv_;
        }
    }
}

// ---------------------------------------------------------------------------
extern "C" void kernel_launch(void* const* d_in, const int* in_sizes, int n_in,
                              void* d_out, int out_size, void* d_ws, size_t ws_size,
                              hipStream_t stream) {
    const float* hidden = (const float*)d_in[0];
    const float* graph  = (const float*)d_in[1];
    const float* Wq = (const float*)d_in[2];  const float* bq = (const float*)d_in[3];
    const float* Wk = (const float*)d_in[4];  const float* bk = (const float*)d_in[5];
    const float* Wv = (const float*)d_in[6];  const float* bv = (const float*)d_in[7];
    const float* Wp = (const float*)d_in[8];  const float* bp = (const float*)d_in[9];
    const float* ml = (const float*)d_in[10];

    float* out      = (float*)d_out;
    float* out_y    = out;                    // 1024*64*256      = 16777216
    float* out_att  = out + 16777216;         // 1024*8*64*64     = 33554432
    float* out_mask = out + 50331648;         // 32768

    char* ws = (char*)d_ws;
    _Float16* W16   = (_Float16*)ws;                     // 4*65536 h = 512 KB
    float*    m_mul = (float*)(ws + 524288);             // 32768 f   = 128 KB
    _Float16* y16   = (_Float16*)(ws + 524288 + 131072); // 16777216 h = 32 MB

    k_convw<<<1024, 256, 0, stream>>>(Wq, Wk, Wv, Wp, W16);
    k_mask <<<128,  256, 0, stream>>>(ml, m_mul, out_mask);
    k_attn_fused<<<8192, 128, 0, stream>>>(hidden, W16, bq, bk, bv,
                                           graph, m_mul, out_att, y16);
    k_proj <<<1024, 128, 0, stream>>>(y16, W16 + 3 * 65536, bp, out_y);
}
